// MultiHeadAttentionLayer_13048110645595
// MI455X (gfx1250) — matrix operand order repaired and checked by reading, not compiled, not run on hardware
//
#include <hip/hip_runtime.h>

#define NHEAD 8
#define DHEAD 64
#define NFEAT 512
#define LWIN  16
#define TTILE 16
#define TLEN  2048
#define BATCH 4

// LDS layout (bytes)
#define SM_WQ    0          // 64*64 bf16 = 8192
#define SM_WK    8192
#define SM_WV    16384
#define SM_BQ    24576      // 64 f32
#define SM_BK    24832
#define SM_BV    25088
#define SM_BO    25344      // 512 f32 -> ends 27392
#define SM_QH    27392      // 16*512 f32 = 32768 -> ends 60160
#define SM_KH    60160      // 8*16*68 f32 = 34816 -> ends 94976
#define SM_VH    94976      // -> ends 129792
#define SM_CT    129792     // 16*512 bf16 = 16384 -> ends 146176
#define SMEM_BYTES 146176
#define KHSTRIDE 68

typedef __attribute__((ext_vector_type(16))) __bf16 v16bf;
typedef __attribute__((ext_vector_type(8)))  float  v8f;

// Native conversion: lets the backend use v_cvt_pk_bf16_f32-class ops instead
// of a 4-VALU software rounding sequence per element.
static __device__ inline __bf16 tobf(float f) { return (__bf16)f; }

// A-fragment (16x32, bf16) from row-major f32 global memory.
// Contiguous k order used here (same as loadB_w/loadB_wo): lane<16 holds row M=lane, K = {0..15}; lane>=16 same row, K = {16..31}.
static __device__ inline v16bf loadA_g(const float* __restrict__ base, int stride, int kb, int lane) {
  const int row = lane & 15, hf = lane >> 4;
  const float* p = base + row * stride + kb + hf * 16;
  float4 x0 = *(const float4*)(p + 0);
  float4 x1 = *(const float4*)(p + 4);
  float4 x2 = *(const float4*)(p + 8);
  float4 x3 = *(const float4*)(p + 12);
  v16bf a;
  a[0]=tobf(x0.x); a[1]=tobf(x0.y); a[2]=tobf(x0.z); a[3]=tobf(x0.w);
  a[4]=tobf(x1.x); a[5]=tobf(x1.y); a[6]=tobf(x1.z); a[7]=tobf(x1.w);
  a[8]=tobf(x2.x); a[9]=tobf(x2.y); a[10]=tobf(x2.z); a[11]=tobf(x2.w);
  a[12]=tobf(x3.x); a[13]=tobf(x3.y); a[14]=tobf(x3.z); a[15]=tobf(x3.w);
  return a;
}

// A-fragment from row-major bf16 LDS.
static __device__ inline v16bf loadA_lds(const __bf16* base, int stride, int kb, int lane) {
  const int row = lane & 15, hf = lane >> 4;
  const __bf16* p = base + row * stride + kb + hf * 16;
  v16bf a;
#pragma unroll
  for (int j = 0; j < 8; ++j) { a[j] = p[j]; a[8 + j] = p[8 + j]; }
  return a;
}

// B-fragment (32x16, bf16) computing X @ W^T: B[K=d][N=e] = W[e][d], W row-major [64][64] bf16 in LDS.
// Contiguous k order used here (same as A loaders): lane holds column N = lane&15; element j -> K = kb + (lane>>4)*16 + j.
static __device__ inline v16bf loadB_w(const __bf16* w, int nb, int kb, int lane) {
  const int c = lane & 15, hf = lane >> 4;
  const __bf16* p = w + (nb + c) * 64 + kb + hf * 16;
  v16bf b;
#pragma unroll
  for (int j = 0; j < 16; ++j) b[j] = p[j];
  return b;
}

// B-fragment for out = ct @ Wo^T: B[K=d][N=e] = Wo[e][d], Wo row-major [512][512] f32 in global (L2-resident).
static __device__ inline v16bf loadB_wo(const float* __restrict__ Wo, int nb, int kb, int lane) {
  const int c = lane & 15, hf = lane >> 4;
  const float* p = Wo + (size_t)(nb + c) * NFEAT + kb + hf * 16;
  float4 x0 = *(const float4*)(p + 0);
  float4 x1 = *(const float4*)(p + 4);
  float4 x2 = *(const float4*)(p + 8);
  float4 x3 = *(const float4*)(p + 12);
  v16bf b;
  b[0]=tobf(x0.x); b[1]=tobf(x0.y); b[2]=tobf(x0.z); b[3]=tobf(x0.w);
  b[4]=tobf(x1.x); b[5]=tobf(x1.y); b[6]=tobf(x1.z); b[7]=tobf(x1.w);
  b[8]=tobf(x2.x); b[9]=tobf(x2.y); b[10]=tobf(x2.z); b[11]=tobf(x2.w);
  b[12]=tobf(x3.x); b[13]=tobf(x3.y); b[14]=tobf(x3.z); b[15]=tobf(x3.w);
  return b;
}

#define WMMA_BF16(A, B, C) \
  __builtin_amdgcn_wmma_f32_16x16x32_bf16(false, (A), false, (B), (short)0, (C), false, false)

__global__ void __launch_bounds__(256)
mha_local_kernel(const float* __restrict__ q, const float* __restrict__ k,
                 const float* __restrict__ v,
                 const float* __restrict__ Wq, const float* __restrict__ bq,
                 const float* __restrict__ Wk, const float* __restrict__ bk,
                 const float* __restrict__ Wv, const float* __restrict__ bv,
                 const float* __restrict__ Wo, const float* __restrict__ bo,
                 float* __restrict__ out)
{
  extern __shared__ char smem[];
  __bf16* wq_s = (__bf16*)(smem + SM_WQ);
  __bf16* wk_s = (__bf16*)(smem + SM_WK);
  __bf16* wv_s = (__bf16*)(smem + SM_WV);
  float*  bq_s = (float*)(smem + SM_BQ);
  float*  bk_s = (float*)(smem + SM_BK);
  float*  bv_s = (float*)(smem + SM_BV);
  float*  bo_s = (float*)(smem + SM_BO);
  float*  qh_s = (float*)(smem + SM_QH);
  float*  kh_s = (float*)(smem + SM_KH);
  float*  vh_s = (float*)(smem + SM_VH);
  __bf16* ct_s = (__bf16*)(smem + SM_CT);

  const int tid  = threadIdx.x;
  const int lane = tid & 31;
  const int wave = tid >> 5;        // wave == head in phases B/C, == 64-col slab in phase D
  const int c    = lane & 15;
  const int hf   = lane >> 4;

  const int b  = blockIdx.x >> 7;            // T/TTILE = 128
  const int tb = (blockIdx.x & 127) << 4;

  // Warm L2 with this wave's Wo slab (used in phase D) while phases A-C run.
  // One prefetch per 128B line; 64 rows x 2KB = 1024 lines / 32 lanes.
  {
    const float* wo_slab = Wo + (size_t)(wave * 64) * NFEAT;
    for (int ln = lane; ln < 1024; ln += 32)
      __builtin_prefetch(wo_slab + ln * 32, 0, 1);
  }

  // ---- Phase A: stage weights + biases to LDS (bf16 weights) ----
  for (int i = tid; i < 64 * 64; i += 256) {
    wq_s[i] = tobf(Wq[i]);
    wk_s[i] = tobf(Wk[i]);
    wv_s[i] = tobf(Wv[i]);
  }
  if (tid < 64) { bq_s[tid] = bq[tid]; bk_s[tid] = bk[tid]; bv_s[tid] = bv[tid]; }
  for (int i = tid; i < NFEAT; i += 256) bo_s[i] = bo[i];
  __syncthreads();

  const int h = wave;

  // ---- Phase B: Q projection, qh = Q_head(16x64) @ Wq^T + bq -> LDS f32 ----
  {
    const float* qbase = q + (size_t)(b * TLEN + tb) * NFEAT + h * DHEAD;   // 16 t-rows, stride NFEAT
    v8f qacc[4] = {};
#pragma unroll
    for (int kc = 0; kc < 2; ++kc) {
      v16bf a = loadA_g(qbase, NFEAT, kc * 32, lane);
#pragma unroll
      for (int nn = 0; nn < 4; ++nn) {
        v16bf bf = loadB_w(wq_s, nn * 16, kc * 32, lane);
        qacc[nn] = WMMA_BF16(a, bf, qacc[nn]);
      }
    }
#pragma unroll
    for (int nn = 0; nn < 4; ++nn) {
      const int   e    = h * DHEAD + nn * 16 + c;
      const float bias = bq_s[nn * 16 + c];
#pragma unroll
      for (int r = 0; r < 8; ++r) {
        const int trow = r + 8 * hf;                  // C-layout: M = r + 8*half
        qh_s[trow * NFEAT + e] = qacc[nn][r] + bias;
      }
    }
  }
  __syncthreads();

  // ---- Phase C: per timestep: K/V projection (WMMA), scores, softmax, context ----
  float* khh = kh_s + h * LWIN * KHSTRIDE;
  float* vhh = vh_s + h * LWIN * KHSTRIDE;

  for (int t = 0; t < TTILE; ++t) {
    const size_t rowbase = (size_t)(b * TLEN + tb + t) * LWIN * NFEAT + h * DHEAD;
    const float* kbase = k + rowbase;   // 16 l-rows, stride NFEAT
    const float* vbase = v + rowbase;

    v8f kacc[4] = {};
    v8f vacc[4] = {};
#pragma unroll
    for (int kc = 0; kc < 2; ++kc) {
      v16bf ka = loadA_g(kbase, NFEAT, kc * 32, lane);
      v16bf va = loadA_g(vbase, NFEAT, kc * 32, lane);
#pragma unroll
      for (int nn = 0; nn < 4; ++nn) {
        v16bf bk_f = loadB_w(wk_s, nn * 16, kc * 32, lane);
        kacc[nn] = WMMA_BF16(ka, bk_f, kacc[nn]);
        v16bf bv_f = loadB_w(wv_s, nn * 16, kc * 32, lane);
        vacc[nn] = WMMA_BF16(va, bv_f, vacc[nn]);
      }
    }
#pragma unroll
    for (int nn = 0; nn < 4; ++nn) {
      const int   e   = nn * 16 + c;
      const float bkv = bk_s[e];
      const float bvv = bv_s[e];
#pragma unroll
      for (int r = 0; r < 8; ++r) {
        const int l = r + 8 * hf;
        khh[l * KHSTRIDE + e] = kacc[nn][r] + bkv;
        vhh[l * KHSTRIDE + e] = vacc[nn][r] + bvv;
      }
    }
    __syncthreads();   // uniform across all 8 waves (same trip count)

    // scores: both lane halves compute l = lane&15 (duplicated -> both halves get softmax)
    const int    l    = lane & 15;
    const float* qrow = qh_s + t * NFEAT + h * DHEAD;
    const float* krow = khh + l * KHSTRIDE;
    float s = 0.f;
#pragma unroll
    for (int e = 0; e < DHEAD; ++e) s += qrow[e] * krow[e];
    s *= 0.125f;   // 1/sqrt(64)

    float m = s;
#pragma unroll
    for (int off = 8; off >= 1; off >>= 1) m = fmaxf(m, __shfl_xor(m, off, 32));
    const float ex = __expf(s - m);
    float sum = ex;
#pragma unroll
    for (int off = 8; off >= 1; off >>= 1) sum += __shfl_xor(sum, off, 32);
    const float p = ex / sum;   // lane j (<16) holds p_j; lanes 16..31 duplicate

    // context: each lane owns 2 output features e = lane*2, lane*2+1
    float ct0 = 0.f, ct1 = 0.f;
#pragma unroll
    for (int l2 = 0; l2 < 16; ++l2) {
      const float  pl = __shfl(p, l2, 32);
      const float2 vv = *(const float2*)(vhh + l2 * KHSTRIDE + lane * 2);
      ct0 += pl * vv.x;
      ct1 += pl * vv.y;
    }
    __bf16* ctp = ct_s + t * NFEAT + h * DHEAD + lane * 2;
    ctp[0] = tobf(ct0);
    ctp[1] = tobf(ct1);
  }
  __syncthreads();   // ct tile is read cross-wave below

  // ---- Phase D: out(16x512) = ct(16x512) @ Wo^T + bo ; wave w owns N-cols [w*64, w*64+64) ----
  {
    v8f oacc[4] = {};
    for (int kc = 0; kc < 16; ++kc) {
      v16bf a = loadA_lds(ct_s, NFEAT, kc * 32, lane);
#pragma unroll
      for (int nt = 0; nt < 4; ++nt) {
        const int nb = wave * 64 + nt * 16;
        v16bf bf = loadB_wo(Wo, nb, kc * 32, lane);
        oacc[nt] = WMMA_BF16(a, bf, oacc[nt]);
      }
    }
    float* obase = out + (size_t)(b * TLEN + tb) * NFEAT;
#pragma unroll
    for (int nt = 0; nt < 4; ++nt) {
      const int   e  = wave * 64 + nt * 16 + c;
      const float bb = bo_s[e];
#pragma unroll
      for (int r = 0; r < 8; ++r) {
        const int trow = r + 8 * hf;
        obase[trow * NFEAT + e] = oacc[nt][r] + bb;
      }
    }
  }
}

extern "C" void kernel_launch(void* const* d_in, const int* in_sizes, int n_in,
                              void* d_out, int out_size, void* d_ws, size_t ws_size,
                              hipStream_t stream) {
  (void)in_sizes; (void)n_in; (void)out_size; (void)d_ws; (void)ws_size;
  const float* q  = (const float*)d_in[0];
  const float* k  = (const float*)d_in[1];
  const float* v  = (const float*)d_in[2];
  const float* Wq = (const float*)d_in[3];
  const float* bq = (const float*)d_in[4];
  const float* Wk = (const float*)d_in[5];
  const float* bk = (const float*)d_in[6];
  const float* Wv = (const float*)d_in[7];
  const float* bv = (const float*)d_in[8];
  const float* Wo = (const float*)d_in[9];
  const float* bo = (const float*)d_in[10];
  float* out = (float*)d_out;

  dim3 grid(BATCH * (TLEN / TTILE));   // 512 workgroups
  dim3 block(256);                     // 8 waves (wave32)
  hipLaunchKernelGGL(mha_local_kernel, grid, block, SMEM_BYTES, stream,
                     q, k, v, Wq, bq, Wk, bk, Wv, bv, Wo, bo, out);
}
